// RNN_17970143166649
// MI455X (gfx1250) — matrix operand order repaired and checked by reading, not compiled, not run on hardware
//
#include <hip/hip_runtime.h>
#include <hip/hip_bf16.h>
#include <math.h>

#define T_STEPS 8192
#define IN_DIM  1024
#define HID_DIM 2048
#define OUT_DIM 256
#define G3      (3*HID_DIM)      /* 6144 */

typedef __bf16 v16bf __attribute__((ext_vector_type(16)));
typedef float  v8f   __attribute__((ext_vector_type(8)));

static __device__ __forceinline__ unsigned short f32_bf16(float f) {
    unsigned u = __float_as_uint(f);
    unsigned r = u + 0x7FFFu + ((u >> 16) & 1u);   // round-to-nearest-even
    return (unsigned short)(r >> 16);
}
static __device__ __forceinline__ float bf16lo_f32(unsigned u) { return __uint_as_float(u << 16); }
static __device__ __forceinline__ float bf16hi_f32(unsigned u) { return __uint_as_float(u & 0xFFFF0000u); }

// ---------------- conversion / packing kernels ----------------

__global__ void k_conv_input(const float* __restrict__ in, unsigned short* __restrict__ a, int n) {
    int i = blockIdx.x * blockDim.x + threadIdx.x;
    if (i < n) a[i] = f32_bf16(in[i]);
}

// Pack w_ih (6144 x 1024, row-major) into per-lane WMMA B-fragment bf16 layout (this code's contiguous k order, matching A):
// tile index = ntile*32 + ktile (ktile over K/32, ntile over 6144/16),
// within tile: lane*16 + i, where element = w_ih[n][k],
//   k = ktile*32 + (lane>>4)*16 + i,  n = ntile*16 + (lane&15)
// so each lane's fragment is 32 contiguous bytes.
__global__ void k_pack_wih(const float* __restrict__ wih, unsigned short* __restrict__ bp) {
    int o = blockIdx.x * blockDim.x + threadIdx.x;
    if (o >= IN_DIM * G3) return;
    int tile   = o >> 9;
    int within = o & 511;
    int lane   = within >> 4;
    int i      = within & 15;
    int ktile  = tile & 31;
    int ntile  = tile >> 5;
    int k = ktile * 32 + ((lane >> 4) << 4) + i;
    int n = ntile * 16 + (lane & 15);
    bp[o] = f32_bf16(wih[n * IN_DIM + k]);
}

// ---------------- WMMA GEMM: igates = A(8192x1024) * W^T + b ----------------
// block = 256 threads (8 waves). Each wave: 16x64 output tile (4 accumulators).
__global__ void __launch_bounds__(256) k_gemm_igates(
        const unsigned short* __restrict__ A,
        const unsigned short* __restrict__ Bp,
        const float* __restrict__ bias,
        float* __restrict__ C) {
    const int lane = threadIdx.x & 31;
    const int wave = threadIdx.x >> 5;
    const int lh = lane & 15;
    const int hi = lane >> 4;
    const int mBase = blockIdx.y * 16;
    const int nBase = blockIdx.x * 512 + wave * 64;

    union Frag { v16bf v; uint4 q[2]; };
    v8f acc[4] = {};

    for (int k0 = 0; k0 < IN_DIM; k0 += 32) {
        Frag af;
        const unsigned short* ap = A + (size_t)(mBase + lh) * IN_DIM + k0 + hi * 16;
        af.q[0] = *(const uint4*)(ap);
        af.q[1] = *(const uint4*)(ap + 8);
#pragma unroll
        for (int s = 0; s < 4; ++s) {
            int ntile = (nBase >> 4) + s;
            const unsigned short* bptr = Bp + ((size_t)(ntile * 32 + (k0 >> 5)) * 512 + lane * 16);
            Frag bf;
            bf.q[0] = *(const uint4*)(bptr);
            bf.q[1] = *(const uint4*)(bptr + 8);
            acc[s] = __builtin_amdgcn_wmma_f32_16x16x32_bf16(
                false, af.v, false, bf.v, (short)0, acc[s], false, false);
        }
    }
#pragma unroll
    for (int s = 0; s < 4; ++s) {
        int n = nBase + s * 16 + lh;
        float bv = bias[n];
        float* cp = C + (size_t)(mBase + hi * 8) * G3 + n;
#pragma unroll
        for (int v = 0; v < 8; ++v) cp[(size_t)v * G3] = acc[s][v] + bv;
    }
}

// ---------------- device-wide barrier (persistent GRU) ----------------

static __device__ __forceinline__ void grid_sync(unsigned* cnt, unsigned* gen, unsigned nb) {
    __syncthreads();
    if (threadIdx.x == 0) {
        __threadfence();
        unsigned g = __hip_atomic_load(gen, __ATOMIC_ACQUIRE, __HIP_MEMORY_SCOPE_AGENT);
        unsigned arrived = __hip_atomic_fetch_add(cnt, 1u, __ATOMIC_ACQ_REL, __HIP_MEMORY_SCOPE_AGENT);
        if (arrived == nb - 1u) {
            __hip_atomic_store(cnt, 0u, __ATOMIC_RELAXED, __HIP_MEMORY_SCOPE_AGENT);
            __hip_atomic_fetch_add(gen, 1u, __ATOMIC_RELEASE, __HIP_MEMORY_SCOPE_AGENT);
        } else {
            while (__hip_atomic_load(gen, __ATOMIC_ACQUIRE, __HIP_MEMORY_SCOPE_AGENT) == g) {
                __builtin_amdgcn_s_sleep(1);
            }
        }
    }
    __syncthreads();
}

// ---------------- persistent GRU recurrence ----------------
// 256 blocks x 256 threads. Block owns hidden indices [blk*8, blk*8+8):
// its 24 w_hh rows (r/z/a) live in LDS as packed bf16 pairs (96 KB) + h mirror (8 KB).
// Wave w computes the 3 dot products for j = blk*8 + w.
__global__ void __launch_bounds__(256, 1) k_gru(
        const float* __restrict__ whh,
        const float* __restrict__ bn,
        const float* __restrict__ ig,
        float* __restrict__ hbuf,
        unsigned* __restrict__ bar) {
    extern __shared__ char smem[];
    unsigned* wsh = (unsigned*)smem;                       // 24 * 1024 uints (bf16 pairs)
    float* h_s = (float*)(smem + 24 * HID_DIM * 2);        // 2048 floats

    const int tid  = threadIdx.x;
    const int lane = tid & 31;
    const int wave = tid >> 5;
    const int blk  = blockIdx.x;

    // stage this block's w_hh slice into LDS as bf16 pairs
    for (int e = tid; e < 24 * 1024; e += 256) {
        int lr   = e >> 10;                // local row 0..23
        int col2 = e & 1023;               // pair index within row
        int gate = lr >> 3, j = lr & 7;
        const float* src = whh + (size_t)(gate * HID_DIM + blk * 8 + j) * HID_DIM + col2 * 2;
        unsigned lo = f32_bf16(src[0]);
        unsigned hh = f32_bf16(src[1]);
        wsh[e] = lo | (hh << 16);
    }
    const int jg = blk * 8 + wave;
    const float bnj = bn[jg];
    if (tid < 8) hbuf[blk * 8 + tid] = 0.0f;   // h0 = 0 (buffer 0)

    grid_sync(bar, bar + 1, gridDim.x);

    const unsigned* wr = wsh + (size_t)(wave)      * 1024;
    const unsigned* wz = wsh + (size_t)(8  + wave) * 1024;
    const unsigned* wa = wsh + (size_t)(16 + wave) * 1024;

    for (int t = 0; t < T_STEPS; ++t) {
        const float* hc = hbuf + (t & 1) * HID_DIM;
        float*       hn = hbuf + ((t + 1) & 1) * HID_DIM;

        for (int i = tid; i < HID_DIM; i += 256) h_s[i] = hc[i];
        __syncthreads();

        float ar = 0.f, az = 0.f, aa = 0.f;
#pragma unroll 4
        for (int i = 0; i < 32; ++i) {
            int u = lane + 32 * i;
            float h0 = h_s[2 * u], h1 = h_s[2 * u + 1];
            unsigned pr = wr[u], pz = wz[u], pa = wa[u];
            ar += bf16lo_f32(pr) * h0 + bf16hi_f32(pr) * h1;
            az += bf16lo_f32(pz) * h0 + bf16hi_f32(pz) * h1;
            aa += bf16lo_f32(pa) * h0 + bf16hi_f32(pa) * h1;
        }
        for (int off = 16; off; off >>= 1) {
            ar += __shfl_down(ar, off, 32);
            az += __shfl_down(az, off, 32);
            aa += __shfl_down(aa, off, 32);
        }
        if (lane == 0) {
            const float* igt = ig + (size_t)t * G3;
            float r = 1.f / (1.f + __expf(-(igt[jg] + ar)));
            float z = 1.f / (1.f + __expf(-(igt[HID_DIM + jg] + az)));
            float n = tanhf(igt[2 * HID_DIM + jg] + r * (aa + bnj));
            hn[jg] = (1.f - z) * n + z * h_s[jg];
        }
        grid_sync(bar, bar + 1, gridDim.x);
    }
}

// ---------------- dropout + LayerNorm + Linear + sigmoid ----------------
__global__ void __launch_bounds__(256) k_head(
        const float* __restrict__ h, const float* __restrict__ mask,
        const float* __restrict__ lnw, const float* __restrict__ lnb,
        const float* __restrict__ linw, const float* __restrict__ linb,
        const float* __restrict__ ob, float* __restrict__ out) {
    __shared__ float xs[HID_DIM];
    __shared__ float red[256], red2[256];
    int tid = threadIdx.x;
    float s = 0.f, s2 = 0.f;
    for (int i = tid; i < HID_DIM; i += 256) {
        float x = h[i] * mask[i];
        xs[i] = x; s += x; s2 += x * x;
    }
    red[tid] = s; red2[tid] = s2;
    __syncthreads();
    for (int o = 128; o; o >>= 1) {
        if (tid < o) { red[tid] += red[tid + o]; red2[tid] += red2[tid + o]; }
        __syncthreads();
    }
    float mu  = red[0] / (float)HID_DIM;
    float var = red2[0] / (float)HID_DIM - mu * mu;
    float inv = rsqrtf(var + 1e-5f);
    __syncthreads();
    for (int i = tid; i < HID_DIM; i += 256) xs[i] = (xs[i] - mu) * inv * lnw[i] + lnb[i];
    __syncthreads();
    float acc = linb[tid] + ob[tid];
    const float* w = linw + (size_t)tid * HID_DIM;
    for (int k = 0; k < HID_DIM; ++k) acc += w[k] * xs[k];
    out[tid] = 1.f / (1.f + __expf(-acc));
}

// ---------------- host launch ----------------
extern "C" void kernel_launch(void* const* d_in, const int* in_sizes, int n_in,
                              void* d_out, int out_size, void* d_ws, size_t ws_size,
                              hipStream_t stream) {
    (void)in_sizes; (void)n_in; (void)out_size; (void)ws_size;
    const float* input   = (const float*)d_in[0];
    const float* w_ih    = (const float*)d_in[1];
    const float* w_hh    = (const float*)d_in[2];
    const float* b       = (const float*)d_in[3];
    const float* b_n     = (const float*)d_in[4];
    const float* mask    = (const float*)d_in[5];
    const float* ln_w    = (const float*)d_in[6];
    const float* ln_b    = (const float*)d_in[7];
    const float* lin_w   = (const float*)d_in[8];
    const float* lin_b   = (const float*)d_in[9];
    const float* out_b   = (const float*)d_in[10];
    float* out = (float*)d_out;

    char* ws = (char*)d_ws;
    unsigned*       bar  = (unsigned*)ws;                          // 256 B
    float*          hbuf = (float*)(ws + 256);                     // 2*2048*4 = 16 KB
    size_t offA = 256 + 2 * HID_DIM * sizeof(float);
    unsigned short* Abf  = (unsigned short*)(ws + offA);           // 16 MB
    size_t szA = (size_t)T_STEPS * IN_DIM * sizeof(unsigned short);
    unsigned short* Bp   = (unsigned short*)(ws + offA + szA);     // 12 MB
    size_t szB = (size_t)IN_DIM * G3 * sizeof(unsigned short);
    float*          ig   = (float*)(ws + offA + szA + szB);        // 201 MB

    hipMemsetAsync(bar, 0, 256, stream);
    k_conv_input<<<(T_STEPS * IN_DIM + 255) / 256, 256, 0, stream>>>(input, Abf, T_STEPS * IN_DIM);
    k_pack_wih<<<(IN_DIM * G3 + 255) / 256, 256, 0, stream>>>(w_ih, Bp);
    k_gemm_igates<<<dim3(G3 / 512, T_STEPS / 16), 256, 0, stream>>>(Abf, Bp, b, ig);

    size_t smem = (size_t)24 * HID_DIM * 2 + (size_t)HID_DIM * 4;  // 104 KB
    hipFuncSetAttribute((const void*)k_gru, hipFuncAttributeMaxDynamicSharedMemorySize, (int)smem);
    k_gru<<<256, 256, smem, stream>>>(w_hh, b_n, ig, hbuf, bar);

    k_head<<<1, 256, 0, stream>>>(hbuf, mask, ln_w, ln_b, lin_w, lin_b, out_b, out);
}
